// RGCN_73289321939190
// MI455X (gfx1250) — compile-verified, matrix operand order checked by reading, not run on hardware
//
#include <hip/hip_runtime.h>

typedef __attribute__((ext_vector_type(16))) __bf16 v16bf;
typedef __attribute__((ext_vector_type(8)))  __bf16 v8bf;
typedef __attribute__((ext_vector_type(8)))  float  v8f;

#define KDIM   128             // IN_DIM == OUT_DIM (K of both GEMMs)
#define NBASES 8
#define XCOLS  (NBASES * KDIM) // 1024
#define KB     (KDIM / 32)     // 4 K-blocks of 32

// ---------- pack: f32 -> bf16 (straight cast, for the A matrix) ----------
__global__ void cast_f32_bf16(const float* __restrict__ in, __bf16* __restrict__ out, int n) {
    for (int i = blockIdx.x * blockDim.x + threadIdx.x; i < n; i += gridDim.x * blockDim.x)
        out[i] = (__bf16)in[i];
}

// ---------- pack B into the WMMA per-lane operand layout ----------
// Dest linear layout: (((tn*KB + kb)*32 + lane)*16 + i) bf16 elements, so each
// lane's 16-element fragment is 32 contiguous bytes (two global_load_b128).
// ISA 16-bit B layout: lane<16 -> K {kb*32+0..7, +16..23}; lane>=16 -> K {+8..15, +24..31};
// vector elem i<8 -> k = kb*32 + half*8 + i ; i>=8 -> k = kb*32 + half*8 + i + 8.
//
// MODE 0 (self_loop): column c (<128)  -> src = k*128 + c
// MODE 1 (weight, b-innermost output): column c (<1024): b = c&7, j = c>>3
//                                      -> src = b*128*128 + k*128 + j
// MODE 1 makes the GEMM emit X[n][j*8+b] directly (edge phase reads v8bf per thread).
template <int MODE>
__global__ void pack_B(const float* __restrict__ srcmat, __bf16* __restrict__ dst, int total) {
    for (int d = blockIdx.x * blockDim.x + threadIdx.x; d < total; d += gridDim.x * blockDim.x) {
        const int tn   = d >> 11;          // 2048 elements per 16-col tile (KB*32*16)
        const int r    = d & 2047;
        const int kb   = r >> 9;           // 512 per K-block
        const int r2   = r & 511;
        const int lane = r2 >> 4;
        const int i    = r2 & 15;
        const int half = lane >> 4;
        const int n    = lane & 15;
        const int k    = kb * 32 + half * 8 + i + ((i >= 8) ? 8 : 0);
        const int c    = tn * 16 + n;
        float v;
        if (MODE == 0) {
            v = srcmat[k * KDIM + c];
        } else {
            const int b = c & 7, j = c >> 3;
            v = srcmat[b * KDIM * KDIM + k * KDIM + j];
        }
        dst[d] = (__bf16)v;
    }
}

// ---------- GEMM: C(M x Ncol) = A(M x 128 bf16) * B(128 x Ncol), f32 accum ----------
// One wave computes a 16x64 output strip: 4 accumulators, 16 WMMAs, A reused 4x.
template <bool BF16OUT>
__global__ void gemm_wmma(const __bf16* __restrict__ A, const v16bf* __restrict__ Bp,
                          float* __restrict__ Cf, __bf16* __restrict__ Cb,
                          int M, int Ncol) {
    const int tilesNG = Ncol >> 6;                         // groups of 4 16-col tiles
    const int tilesM  = (M + 15) >> 4;
    const int wid = blockIdx.x * (blockDim.x >> 5) + (threadIdx.x >> 5);
    if (wid >= tilesM * tilesNG) return;                   // uniform per-wave exit
    const int tm = wid / tilesNG;
    const int tg = wid - tm * tilesNG;

    const int lane = threadIdx.x & 31;
    const int half = lane >> 4;
    const int lr   = lane & 15;

    int arow = tm * 16 + lr; if (arow >= M) arow = M - 1;  // clamp: EXEC stays full
    const __bf16* abase = A + (size_t)arow * KDIM + half * 8;

    v8f c0 = {}, c1 = {}, c2 = {}, c3 = {};
    #pragma unroll
    for (int kb = 0; kb < KB; ++kb) {
        const v8bf lo = *(const v8bf*)(abase + kb * 32);
        const v8bf hi = *(const v8bf*)(abase + kb * 32 + 16);
        const v16bf a = __builtin_shufflevector(lo, hi, 0, 1, 2, 3, 4, 5, 6, 7,
                                                8, 9, 10, 11, 12, 13, 14, 15);
        const size_t base = ((size_t)(tg * 4) * KB + kb) * 32 + lane;
        const v16bf b0 = Bp[base];
        const v16bf b1 = Bp[base + (size_t)KB * 32];
        const v16bf b2 = Bp[base + (size_t)2 * KB * 32];
        const v16bf b3 = Bp[base + (size_t)3 * KB * 32];
        c0 = __builtin_amdgcn_wmma_f32_16x16x32_bf16(false, a, false, b0, (short)0, c0, false, false);
        c1 = __builtin_amdgcn_wmma_f32_16x16x32_bf16(false, a, false, b1, (short)0, c1, false, false);
        c2 = __builtin_amdgcn_wmma_f32_16x16x32_bf16(false, a, false, b2, (short)0, c2, false, false);
        c3 = __builtin_amdgcn_wmma_f32_16x16x32_bf16(false, a, false, b3, (short)0, c3, false, false);
    }

    // C/D layout: VGPR r -> M = half*8 + r, N = lane&15
    #pragma unroll
    for (int r = 0; r < 8; ++r) {
        const int row = tm * 16 + half * 8 + r;
        if (row < M) {
            const size_t rb = (size_t)row * Ncol + tg * 64 + lr;
            if (BF16OUT) {
                Cb[rb]      = (__bf16)c0[r];
                Cb[rb + 16] = (__bf16)c1[r];
                Cb[rb + 32] = (__bf16)c2[r];
                Cb[rb + 48] = (__bf16)c3[r];
            } else {
                Cf[rb]      = c0[r];
                Cf[rb + 16] = c1[r];
                Cf[rb + 32] = c2[r];
                Cf[rb + 48] = c3[r];
            }
        }
    }
}

// ---------- edge scatter: out[dst][j] += norm * sum_b coef[et][b] * X[src][j*8+b] ----------
__global__ void edge_scatter(const __bf16* __restrict__ X, const float* __restrict__ w_coe,
                             const float* __restrict__ norm, const int* __restrict__ src,
                             const int* __restrict__ dst, const int* __restrict__ etype,
                             float* __restrict__ out, int E) {
    const int e = blockIdx.x * 2 + (threadIdx.x >> 7);
    if (e >= E) return;
    const int j = threadIdx.x & 127;

    const int   s  = src[e];
    const int   d  = dst[e];
    const int   t  = etype[e];
    const float nm = norm[e];

    const v8bf   x  = *(const v8bf*)(X + (size_t)s * XCOLS + j * 8);   // one b128 load
    const float4* cp = (const float4*)(w_coe + t * NBASES);            // 32B aligned
    const float4 cA = cp[0], cB = cp[1];

    float y = cA.x * (float)x[0] + cA.y * (float)x[1] +
              cA.z * (float)x[2] + cA.w * (float)x[3] +
              cB.x * (float)x[4] + cB.y * (float)x[5] +
              cB.z * (float)x[6] + cB.w * (float)x[7];

    atomicAdd(out + (size_t)d * KDIM + j, y * nm);
}

static inline size_t align256(size_t x) { return (x + 255) & ~(size_t)255; }

extern "C" void kernel_launch(void* const* d_in, const int* in_sizes, int n_in,
                              void* d_out, int out_size, void* d_ws, size_t ws_size,
                              hipStream_t stream) {
    const float* node_feat = (const float*)d_in[0];
    const float* weight    = (const float*)d_in[1];
    const float* w_coe     = (const float*)d_in[2];
    const float* self_loop = (const float*)d_in[3];
    const float* norm      = (const float*)d_in[4];
    const int*   src       = (const int*)d_in[5];
    const int*   dst       = (const int*)d_in[6];
    const int*   etype     = (const int*)d_in[7];

    const int Nn = in_sizes[0] / KDIM;
    const int E  = in_sizes[5];

    // workspace carve-up
    char* ws = (char*)d_ws;
    size_t off = 0;
    __bf16* X   = (__bf16*)(ws + off); off += align256((size_t)Nn * XCOLS * 2);
    __bf16* A16 = (__bf16*)(ws + off); off += align256((size_t)Nn * KDIM * 2);
    __bf16* B1  = (__bf16*)(ws + off); off += align256((size_t)KDIM * XCOLS * 2);
    __bf16* B2  = (__bf16*)(ws + off); off += align256((size_t)KDIM * KDIM * 2);
    (void)ws_size;

    // 1) pack operands: A -> bf16 row-major; B1/B2 -> WMMA lane layout
    {
        int n = Nn * KDIM;
        cast_f32_bf16<<<(n + 255) / 256, 256, 0, stream>>>(node_feat, A16, n);
        pack_B<0><<<(KDIM * KDIM + 255) / 256, 256, 0, stream>>>(self_loop, B2, KDIM * KDIM);
        pack_B<1><<<(KDIM * XCOLS + 255) / 256, 256, 0, stream>>>(weight, B1, KDIM * XCOLS);
    }

    // 2) out = node_feat @ self_loop  (f32, fully initializes d_out)
    {
        int waves  = ((Nn + 15) / 16) * (KDIM / 64);
        int blocks = (waves + 3) / 4;                     // 4 waves per 128-thread block
        gemm_wmma<false><<<blocks, 128, 0, stream>>>(A16, (const v16bf*)B2,
                                                     (float*)d_out, nullptr, Nn, KDIM);
    }

    // 3) X[n][j*8+b] = (node_feat @ W_b)[n][j]  (bf16 intermediate, N x 1024, b innermost)
    {
        int waves  = ((Nn + 15) / 16) * (XCOLS / 64);
        int blocks = (waves + 3) / 4;
        gemm_wmma<true><<<blocks, 128, 0, stream>>>(A16, (const v16bf*)B1,
                                                    nullptr, X, Nn, XCOLS);
    }

    // 4) edge gather (one b128/thread) + atomic scatter into L2-resident out
    {
        int blocks = (E + 1) / 2;                         // 2 edges per 256-thread block
        edge_scatter<<<blocks, 256, 0, stream>>>(X, w_coe, norm, src, dst, etype,
                                                 (float*)d_out, E);
    }
}
